// TransformerBlock_15556371546288
// MI455X (gfx1250) — compile-verified, matrix operand order checked by reading, not run on hardware
//
#include <hip/hip_runtime.h>

typedef __attribute__((ext_vector_type(16))) __bf16 v16bf;
typedef __attribute__((ext_vector_type(8)))  float  v8f;

#define B_ 4
#define T_ 2048
#define K_ 256
#define H_ 8
#define M_ (B_*T_)    /* 8192 rows (b*T+t) */
#define NQ (K_*H_)    /* 2048 */
#define NF (4*K_)     /* 1024 */

union Frag {
  v16bf bf;
  uint4 q[2];
  unsigned short u[16];
};

__device__ __forceinline__ unsigned short f2bf(float f){
  unsigned int u = __float_as_uint(f);
  unsigned int r = u + 0x7FFFu + ((u >> 16) & 1u);   // round-to-nearest-even
  return (unsigned short)(r >> 16);
}

// Load a 16x32 bf16 fragment (A-layout; B-operands are stored transposed and
// use the identical per-lane pattern): lane L holds row (L&15); lanes 0-15
// take k-offsets {0..7,16..23}, lanes 16-31 take {8..15,24..31}.
__device__ __forceinline__ void load_frag(Frag& f, const unsigned short* base,
                                          int strideElems, int lane){
  int r  = lane & 15;
  int ko = (lane & 16) ? 8 : 0;
  const unsigned short* p = base + (long)r * strideElems + ko;
  f.q[0] = *reinterpret_cast<const uint4*>(p);
  f.q[1] = *reinterpret_cast<const uint4*>(p + 16);
}

__device__ __forceinline__ v8f wmma_bf(const Frag& a, const Frag& b, v8f c){
  return __builtin_amdgcn_wmma_f32_16x16x32_bf16(false, a.bf, false, b.bf,
                                                 (short)0, c, false, false);
}

__global__ void k_f32_to_bf16(const float* __restrict__ in,
                              unsigned short* __restrict__ out, int n){
  int i = blockIdx.x * 256 + threadIdx.x;
  if (i < n) out[i] = f2bf(in[i]);
}

// in: f32 [R][C] row-major  ->  out: bf16 [C][R]
__global__ void k_transpose_bf16(const float* __restrict__ in,
                                 unsigned short* __restrict__ out, int R, int C){
  int i = blockIdx.x * 256 + threadIdx.x;
  if (i < R * C){
    int r = i / C, c = i - r * C;
    out[(long)c * R + r] = f2bf(in[i]);
  }
}

// Generic WMMA GEMM: C[16x64 per wave] = A[M x Kin] * Bt[N x Kin]^T
// MODE 0: q/k projection -> bf16 [b,h,t,d], scaled by 0.25 (= 256^-0.25)
// MODE 1: v projection   -> bf16 [b,h,d,t] (transposed for PV B-operand)
// MODE 2: f32 out = acc + bias[n] + resid[m*N+n]
// MODE 3: bf16 out = relu(acc + bias[n])
template<int MODE>
__global__ void __launch_bounds__(256)
k_gemm(const unsigned short* __restrict__ A,
       const unsigned short* __restrict__ Bt,
       const float* __restrict__ bias,
       const float* __restrict__ resid,
       float* __restrict__ outF,
       unsigned short* __restrict__ outB,
       int N, int Kin)
{
  int lane = threadIdx.x & 31;
  int w    = blockIdx.x * 8 + (threadIdx.x >> 5);
  int ntg  = N >> 6;
  int mi   = w / ntg, ni = w - mi * ntg;
  long m0  = (long)mi * 16, n0 = (long)ni * 64;

  v8f z = {0.f,0.f,0.f,0.f,0.f,0.f,0.f,0.f};
  v8f acc[4] = {z, z, z, z};

  const unsigned short* arow = A + m0 * Kin;
  for (int k0 = 0; k0 < Kin; k0 += 32){
    if (k0 + 32 < Kin)   // stage next A chunk into cache while WMMAs run
      __builtin_prefetch(arow + (long)(lane & 15) * Kin + k0 + 32, 0, 3);
    Frag af; load_frag(af, arow + k0, Kin, lane);
#pragma unroll
    for (int j = 0; j < 4; ++j){
      Frag bf; load_frag(bf, Bt + (n0 + 16*j) * Kin + k0, Kin, lane);
      acc[j] = wmma_bf(af, bf, acc[j]);
    }
  }

  int col    = lane & 15;
  int rowoff = (lane & 16) ? 8 : 0;
#pragma unroll
  for (int j = 0; j < 4; ++j){
#pragma unroll
    for (int g = 0; g < 8; ++g){
      long m = m0 + g + rowoff;
      long n = n0 + 16*j + col;
      float v = acc[j][g];
      if constexpr (MODE == 0){
        long b = m >> 11, t = m & 2047;
        long h = n >> 8,  kd = n & 255;
        outB[(((b*H_ + h)*T_ + t) << 8) + kd] = f2bf(v * 0.25f);
      } else if constexpr (MODE == 1){
        long b = m >> 11, t = m & 2047;
        long h = n >> 8,  kd = n & 255;
        outB[((((b*H_ + h) << 8) + kd) * T_) + t] = f2bf(v);
      } else if constexpr (MODE == 2){
        outF[m * N + n] = v + bias[n] + resid[m * N + n];
      } else {
        float y = v + bias[n];
        outB[m * N + n] = f2bf(fmaxf(y, 0.f));
      }
    }
  }
}

// Flash attention: one wave owns 16 query rows of one (b,h); online softmax.
// Q: bf16 [bh][t][256] (pre-scaled), Km: bf16 [bh][s][256] (pre-scaled),
// Vt: bf16 [bh][256][t], O: bf16 [b*T+t][h*256+d]
__global__ void __launch_bounds__(256)
k_flash(const unsigned short* __restrict__ Q,
        const unsigned short* __restrict__ Km,
        const unsigned short* __restrict__ Vt,
        unsigned short* __restrict__ O)
{
  __shared__ __align__(16) unsigned short pbuf[8][16*32];  // per-wave P tile
  int lane = threadIdx.x & 31;
  int wid  = threadIdx.x >> 5;
  int wg   = blockIdx.x * 8 + wid;
  int tt   = wg & 127;          // t-tile within head
  int bh   = wg >> 7;           // 0..31
  int tbase = tt << 4;
  int col    = lane & 15;
  int rowoff = (lane & 16) ? 8 : 0;

  Frag qf[8];
  long qb = ((long)bh * T_ + tbase) << 8;
#pragma unroll
  for (int c = 0; c < 8; ++c) load_frag(qf[c], Q + qb + 32*c, K_, lane);

  v8f z = {0.f,0.f,0.f,0.f,0.f,0.f,0.f,0.f};
  v8f oacc[16];
#pragma unroll
  for (int n = 0; n < 16; ++n) oacc[n] = z;
  float rmax[8], rsum[8];
#pragma unroll
  for (int g = 0; g < 8; ++g){ rmax[g] = -1e30f; rsum[g] = 0.f; }

  int tmaxr = tbase + 15;
  long vb = ((long)bh << 8) * T_;
  for (int s0 = 0; s0 <= tmaxr; s0 += 32){
    long kb = ((long)bh * T_ + s0) << 8;
    if (s0 + 32 <= tmaxr){
      // stage next K chunk (32 rows x 512B) and next V columns into L2/L0
      __builtin_prefetch(Km + kb + (32 << 8) + (long)lane * K_, 0, 3);
      __builtin_prefetch(Vt + vb + (long)lane * 8 * T_ + s0 + 32, 0, 3);
    }
    v8f s1 = z, s2 = z;
#pragma unroll
    for (int c = 0; c < 8; ++c){
      Frag kf;  load_frag(kf,  Km + kb + 32*c,              K_, lane);
      s1 = wmma_bf(qf[c], kf, s1);
      Frag kf2; load_frag(kf2, Km + kb + (16 << 8) + 32*c,  K_, lane);
      s2 = wmma_bf(qf[c], kf2, s2);
    }

    bool needMask = (s0 + 31 > tbase);
    float fac[8];
#pragma unroll
    for (int g = 0; g < 8; ++g){
      float a = s1[g], b = s2[g];
      if (needMask){
        int t = tbase + g + rowoff;
        if (s0 + col > t)      a = -1e30f;
        if (s0 + 16 + col > t) b = -1e30f;
      }
      float mloc = fmaxf(a, b);
#pragma unroll
      for (int mm = 8; mm >= 1; mm >>= 1)
        mloc = fmaxf(mloc, __shfl_xor(mloc, mm, 32));
      float mnew = fmaxf(rmax[g], mloc);
      float f  = __expf(rmax[g] - mnew);
      float p1 = __expf(a - mnew);
      float p2 = __expf(b - mnew);
      float ps = p1 + p2;
#pragma unroll
      for (int mm = 8; mm >= 1; mm >>= 1) ps += __shfl_xor(ps, mm, 32);
      rsum[g] = rsum[g] * f + ps;
      rmax[g] = mnew;
      fac[g]  = f;
      pbuf[wid][(g + rowoff)*32 + col]      = f2bf(p1);
      pbuf[wid][(g + rowoff)*32 + col + 16] = f2bf(p2);
    }

#pragma unroll
    for (int n = 0; n < 16; ++n)
#pragma unroll
      for (int g = 0; g < 8; ++g)
        oacc[n][g] *= fac[g];

    Frag pf; load_frag(pf, &pbuf[wid][0], 32, lane);  // LDS in-order per wave
#pragma unroll
    for (int n = 0; n < 16; ++n){
      Frag vf; load_frag(vf, Vt + vb + (long)(n*16) * T_ + s0, T_, lane);
      oacc[n] = wmma_bf(pf, vf, oacc[n]);
    }
  }

  int b = bh >> 3, h = bh & 7;
#pragma unroll
  for (int g = 0; g < 8; ++g){
    float inv = 1.f / rsum[g];
    long t = tbase + g + rowoff;
    long rowbase = ((long)b * T_ + t) * NQ + (long)h * K_;
#pragma unroll
    for (int n = 0; n < 16; ++n)
      O[rowbase + n*16 + col] = f2bf(oacc[n][g] * inv);
  }
}

// Row LayerNorm over K_=256; block = 256 threads = one row.
__global__ void __launch_bounds__(256)
k_layernorm(const float* __restrict__ in,
            const float* __restrict__ ga,
            const float* __restrict__ be,
            float* __restrict__ outF,
            unsigned short* __restrict__ outB)
{
  __shared__ float sm1[8], sm2[8];
  int r = blockIdx.x, tid = threadIdx.x;
  int lane = tid & 31, wid = tid >> 5;
  float v = in[(long)r * K_ + tid];

  float s = v;
#pragma unroll
  for (int mm = 16; mm >= 1; mm >>= 1) s += __shfl_xor(s, mm, 32);
  if (lane == 0) sm1[wid] = s;
  __syncthreads();
  float tot = 0.f;
#pragma unroll
  for (int i = 0; i < 8; ++i) tot += sm1[i];
  float mean = tot * (1.f / K_);

  float d = v - mean;
  float sq = d * d;
#pragma unroll
  for (int mm = 16; mm >= 1; mm >>= 1) sq += __shfl_xor(sq, mm, 32);
  if (lane == 0) sm2[wid] = sq;
  __syncthreads();
  float tot2 = 0.f;
#pragma unroll
  for (int i = 0; i < 8; ++i) tot2 += sm2[i];
  float var = tot2 * (1.f / K_);

  float y = d * rsqrtf(var + 1e-5f) * ga[tid] + be[tid];
  outF[(long)r * K_ + tid] = y;
  if (outB) outB[(long)r * K_ + tid] = f2bf(y);
}

extern "C" void kernel_launch(void* const* d_in, const int* in_sizes, int n_in,
                              void* d_out, int out_size, void* d_ws, size_t ws_size,
                              hipStream_t stream)
{
  const float* x   = (const float*)d_in[0];
  const float* wq  = (const float*)d_in[1];
  const float* wk  = (const float*)d_in[2];
  const float* wv  = (const float*)d_in[3];
  const float* wu  = (const float*)d_in[4];
  const float* bu  = (const float*)d_in[5];
  const float* w1  = (const float*)d_in[6];
  const float* b1  = (const float*)d_in[7];
  const float* w2  = (const float*)d_in[8];
  const float* b2  = (const float*)d_in[9];
  const float* g1  = (const float*)d_in[10];
  const float* be1 = (const float*)d_in[11];
  const float* g2  = (const float*)d_in[12];
  const float* be2 = (const float*)d_in[13];
  (void)in_sizes; (void)n_in; (void)out_size; (void)ws_size;

  char* base = (char*)d_ws;
  size_t off = 0;
  auto carve = [&](size_t bytes) -> void* {
    void* p = base + off;
    off += (bytes + 255) & ~(size_t)255;
    return p;
  };
  unsigned short* xbf = (unsigned short*)carve((size_t)M_ * K_ * 2);
  unsigned short* wqT = (unsigned short*)carve((size_t)NQ * K_ * 2);
  unsigned short* wkT = (unsigned short*)carve((size_t)NQ * K_ * 2);
  unsigned short* wvT = (unsigned short*)carve((size_t)NQ * K_ * 2);
  unsigned short* wuT = (unsigned short*)carve((size_t)K_ * NQ * 2);
  unsigned short* w1T = (unsigned short*)carve((size_t)NF * K_ * 2);
  unsigned short* w2T = (unsigned short*)carve((size_t)K_ * NF * 2);
  unsigned short* qb  = (unsigned short*)carve((size_t)B_ * H_ * T_ * K_ * 2);
  unsigned short* kb  = (unsigned short*)carve((size_t)B_ * H_ * T_ * K_ * 2);
  unsigned short* vt  = (unsigned short*)carve((size_t)B_ * H_ * T_ * K_ * 2);
  unsigned short* ob  = (unsigned short*)carve((size_t)M_ * NQ * 2);
  unsigned short* ffb = (unsigned short*)carve((size_t)M_ * NF * 2);
  unsigned short* y1b = (unsigned short*)carve((size_t)M_ * K_ * 2);
  float* h1  = (float*)carve((size_t)M_ * K_ * 4);   // reused as h2 after LN1
  float* y1f = (float*)carve((size_t)M_ * K_ * 4);

  // --- precision conversion / weight transposes (bf16, Bt = W^T row-major)
  k_f32_to_bf16<<<(M_*K_ + 255)/256, 256, 0, stream>>>(x, xbf, M_*K_);
  k_transpose_bf16<<<(K_*NQ + 255)/256, 256, 0, stream>>>(wq, wqT, K_, NQ);
  k_transpose_bf16<<<(K_*NQ + 255)/256, 256, 0, stream>>>(wk, wkT, K_, NQ);
  k_transpose_bf16<<<(K_*NQ + 255)/256, 256, 0, stream>>>(wv, wvT, K_, NQ);
  k_transpose_bf16<<<(NQ*K_ + 255)/256, 256, 0, stream>>>(wu, wuT, NQ, K_);
  k_transpose_bf16<<<(K_*NF + 255)/256, 256, 0, stream>>>(w1, w1T, K_, NF);
  k_transpose_bf16<<<(NF*K_ + 255)/256, 256, 0, stream>>>(w2, w2T, NF, K_);

  // --- QKV projections (WMMA)
  int qkvBlocks = (M_/16) * (NQ/64) / 8;   // 2048
  k_gemm<0><<<qkvBlocks, 256, 0, stream>>>(xbf, wqT, nullptr, nullptr, nullptr, qb, NQ, K_);
  k_gemm<0><<<qkvBlocks, 256, 0, stream>>>(xbf, wkT, nullptr, nullptr, nullptr, kb, NQ, K_);
  k_gemm<1><<<qkvBlocks, 256, 0, stream>>>(xbf, wvT, nullptr, nullptr, nullptr, vt, NQ, K_);

  // --- causal flash attention (WMMA QK^T + PV, online softmax)
  k_flash<<<(B_*H_*(T_/16)) / 8, 256, 0, stream>>>(qb, kb, vt, ob);

  // --- output projection + bias + residual (f32)
  k_gemm<2><<<(M_/16)*(K_/64)/8, 256, 0, stream>>>(ob, wuT, bu, x, h1, nullptr, K_, NQ);

  // --- LayerNorm 1 (f32 + bf16 copy for FF)
  k_layernorm<<<M_, 256, 0, stream>>>(h1, g1, be1, y1f, y1b);

  // --- FF1: relu(y1 @ w1 + b1) -> bf16
  k_gemm<3><<<(M_/16)*(NF/64)/8, 256, 0, stream>>>(y1b, w1T, b1, nullptr, nullptr, ffb, NF, K_);

  // --- FF2: y1 + ff @ w2 + b2 -> f32 (reuse h1)
  k_gemm<2><<<(M_/16)*(K_/64)/8, 256, 0, stream>>>(ffb, w2T, b2, y1f, h1, nullptr, K_, NF);

  // --- LayerNorm 2 -> final output
  k_layernorm<<<M_, 256, 0, stream>>>(h1, g2, be2, (float*)d_out, nullptr);
}
